// MultiHeadAttentionQuantum_65481071407411
// MI455X (gfx1250) — compile-verified, matrix operand order checked by reading, not run on hardware
//
#include <hip/hip_runtime.h>
#include <hip/hip_bf16.h>

#define EDIM 1024            // K == N == embed dim
#define BM 128
#define BN 128
#define BK 64
#define NKT (EDIM / BK)      // 16 K-stages
#define LSTRIDE 72           // halfs per LDS row: 64 + 8 pad -> 144B (16B aligned, bank-conflict-free)
#define THREADS 256

typedef _Float16 v4h  __attribute__((ext_vector_type(4)));
typedef _Float16 v8h  __attribute__((ext_vector_type(8)));
typedef _Float16 v16h __attribute__((ext_vector_type(16)));
typedef float    v4f  __attribute__((ext_vector_type(4)));
typedef float    v8f  __attribute__((ext_vector_type(8)));

__global__ __launch_bounds__(THREADS)
void qmha_cos_gemm_wmma(const float* __restrict__ x,     // [M, EDIM]
                        const float* __restrict__ rx,    // [32]
                        const float* __restrict__ W,     // [EDIM, EDIM] (row n is K-contiguous)
                        const float* __restrict__ bias,  // [EDIM]
                        float* __restrict__ out)         // [M, EDIM]
{
    __shared__ _Float16 ldsA[2][BM * LSTRIDE];
    __shared__ _Float16 ldsB[2][BN * LSTRIDE];

    const int tid  = threadIdx.x;
    const int lane = tid & 31;
    const int wave = tid >> 5;       // 0..7
    const int wm   = wave >> 2;      // 0..1 : 64-row group
    const int wn   = wave & 3;       // 0..3 : 32-col group
    const int r    = lane & 15;      // tile row within fragment
    const int hl   = lane >> 4;      // half-wave select (K-offset split)

    const int  nBase = blockIdx.x * BN;
    const long mBase = (long)blockIdx.y * BM;

    // ---- loader indexing: 16 threads per row (float4 each), 16 rows/pass, 8 passes ----
    const int c16  = tid & 15;
    const int r0   = tid >> 4;
    const int kcol = c16 * 4;
    // k_global & 31 == kcol+c & 31 for every stage (BK % 32 == 0) -> loop-invariant phases
    const float phi0 = rx[(kcol + 0) & 31];
    const float phi1 = rx[(kcol + 1) & 31];
    const float phi2 = rx[(kcol + 2) & 31];
    const float phi3 = rx[(kcol + 3) & 31];

    v8f acc[4][2];
#pragma unroll
    for (int i = 0; i < 4; ++i)
#pragma unroll
        for (int j = 0; j < 2; ++j) {
            v8f z = {0.f, 0.f, 0.f, 0.f, 0.f, 0.f, 0.f, 0.f};
            acc[i][j] = z;
        }

    v4f xr[8], wr[8];   // global-fetch staging registers (held across the WMMA block)

    auto fetch = [&](int kt) {
        const int k = kt * BK + kcol;
#pragma unroll
        for (int i = 0; i < 8; ++i) {
            const int rr = r0 + i * 16;
            xr[i] = *(const v4f*)(x + (mBase + (long)rr) * EDIM + k);
            wr[i] = *(const v4f*)(W + (size_t)(nBase + rr) * EDIM + k);
        }
    };

    // Deep prefetch: 2 stages ahead, one global_prefetch_b8 per thread per tile.
    // Each 128x64-f32 tile = 256 x 128B cachelines; 256 threads cover it exactly:
    // thread t -> row t>>1, line half t&1.
    auto prefetch = [&](int kt) {
        const int prow = tid >> 1;
        const int pk   = kt * BK + (tid & 1) * 32;
        __builtin_prefetch(x + (mBase + (long)prow) * EDIM + pk, 0, 3);
        __builtin_prefetch(W + (size_t)(nBase + prow) * EDIM + pk, 0, 3);
    };

    auto stage = [&](int buf) {
        _Float16* As = &ldsA[buf][0];
        _Float16* Bs = &ldsB[buf][0];
#pragma unroll
        for (int i = 0; i < 8; ++i) {
            const int rr = r0 + i * 16;
            v4f a = xr[i];
            v4h ah;
            ah.x = (_Float16)__cosf(a.x + phi0);   // RX(x)·RX(phi)|0> -> <Z> = cos(x+phi)
            ah.y = (_Float16)__cosf(a.y + phi1);
            ah.z = (_Float16)__cosf(a.z + phi2);
            ah.w = (_Float16)__cosf(a.w + phi3);
            *(v4h*)(As + rr * LSTRIDE + kcol) = ah;
            v4f w4 = wr[i];
            v4h wh;
            wh.x = (_Float16)w4.x;
            wh.y = (_Float16)w4.y;
            wh.z = (_Float16)w4.z;
            wh.w = (_Float16)w4.w;
            *(v4h*)(Bs + rr * LSTRIDE + kcol) = wh;
        }
    };

    auto compute = [&](int buf) {
        const _Float16* As = &ldsA[buf][0];
        const _Float16* Bs = &ldsB[buf][0];
#pragma unroll
        for (int kk = 0; kk < BK; kk += 32) {
            v16h afrag[4];
            v16h bfrag[2];
            // 16-bit A/B fragment layout: lane<16 holds K[kk..kk+7] + K[kk+16..kk+23],
            // lane>=16 holds K[kk+8..kk+15] + K[kk+24..kk+31]  -> two ds_load_b128 per lane
#pragma unroll
            for (int i = 0; i < 4; ++i) {
                const _Float16* p = As + (wm * 64 + i * 16 + r) * LSTRIDE + kk + hl * 8;
                v8h lo = *(const v8h*)p;
                v8h hi = *(const v8h*)(p + 16);
                afrag[i] = __builtin_shufflevector(lo, hi,
                    0,1,2,3,4,5,6,7,8,9,10,11,12,13,14,15);
            }
#pragma unroll
            for (int j = 0; j < 2; ++j) {
                const _Float16* p = Bs + (wn * 32 + j * 16 + r) * LSTRIDE + kk + hl * 8;
                v8h lo = *(const v8h*)p;
                v8h hi = *(const v8h*)(p + 16);
                bfrag[j] = __builtin_shufflevector(lo, hi,
                    0,1,2,3,4,5,6,7,8,9,10,11,12,13,14,15);
            }
#pragma unroll
            for (int i = 0; i < 4; ++i)
#pragma unroll
                for (int j = 0; j < 2; ++j)
                    acc[i][j] = __builtin_amdgcn_wmma_f32_16x16x32_f16(
                        false, afrag[i], false, bfrag[j],
                        (short)0, acc[i][j], false, false);
        }
    };

    // ---- software pipeline: double-buffered LDS, one barrier per K-stage ----
    fetch(0);
    prefetch(1);
    stage(0);
    for (int kt = 0; kt < NKT; ++kt) {
        __syncthreads();
        if (kt + 1 < NKT) fetch(kt + 1);      // global loads issued before WMMA block
        if (kt + 2 < NKT) prefetch(kt + 2);   // global_prefetch_b8, 2 stages ahead
        compute(kt & 1);
        if (kt + 1 < NKT) stage((kt + 1) & 1);
    }

    // ---- epilogue: C/D layout (VGPR v: lane<16 -> M=v, lane>=16 -> M=v+8) + bias ----
#pragma unroll
    for (int j = 0; j < 2; ++j) {
        const int n  = nBase + wn * 32 + j * 16 + r;
        const float bv = bias[n];
#pragma unroll
        for (int i = 0; i < 4; ++i) {
            const long m0 = mBase + wm * 64 + i * 16 + hl * 8;
#pragma unroll
            for (int v = 0; v < 8; ++v)
                out[(m0 + v) * EDIM + n] = acc[i][j][v] + bv;
        }
    }
}

extern "C" void kernel_launch(void* const* d_in, const int* in_sizes, int n_in,
                              void* d_out, int out_size, void* d_ws, size_t ws_size,
                              hipStream_t stream) {
    const float* x    = (const float*)d_in[0];   // [8,4096,1024] f32
    const float* rx   = (const float*)d_in[1];   // [32] f32
    const float* W    = (const float*)d_in[2];   // [1024,1024] f32
    const float* bias = (const float*)d_in[3];   // [1024] f32
    float* out        = (float*)d_out;           // [8,4096,1024] f32

    const int Mtot = in_sizes[0] / EDIM;         // 32768
    dim3 grid(EDIM / BN, Mtot / BM);             // (8, 256)
    qmha_cos_gemm_wmma<<<grid, THREADS, 0, stream>>>(x, rx, W, bias, out);
}
